// MultiHeadAttention_24352464570031
// MI455X (gfx1250) — matrix operand order repaired and checked by reading, not compiled, not run on hardware
//
#include <hip/hip_runtime.h>
#include <hip/hip_bf16.h>

typedef __attribute__((ext_vector_type(16))) _Float16 v16h;
typedef __attribute__((ext_vector_type(8)))  _Float16 v8h;
typedef __attribute__((ext_vector_type(4)))  _Float16 v4h;
typedef __attribute__((ext_vector_type(8)))  float    v8f;
typedef __attribute__((ext_vector_type(4)))  float    v4f;
typedef __attribute__((ext_vector_type(4)))  unsigned int ui32x4;
typedef __attribute__((ext_vector_type(4)))  int      i32x4;
typedef __attribute__((ext_vector_type(8)))  int      i32x8;

constexpr int B_  = 4;
constexpr int S_  = 2048;
constexpr int D_  = 1024;
constexpr int H_  = 16;
constexpr int DK_ = 64;
constexpr int NROW_ = B_ * S_;        // 8192
constexpr float SCALE_ = 0.125f;      // 1/sqrt(64)

static __device__ __forceinline__ v8f wmma_f16(v16h a, v16h b, v8f c) {
  return __builtin_amdgcn_wmma_f32_16x16x32_f16(false, a, false, b, (short)0, c,
                                                false, false);
}

// A-fragment (16-bit, 16x32): lane half selects K groups {0-7,16-23}/{8-15,24-31}
static __device__ __forceinline__ v16h load_a_frag(const _Float16* p) {
  v8h lo = *(const v8h*)(p);        // K = base .. base+7
  v8h hi = *(const v8h*)(p + 16);   // K = base+16 .. base+23
  return __builtin_shufflevector(lo, hi, 0, 1, 2, 3, 4, 5, 6, 7,
                                 8, 9, 10, 11, 12, 13, 14, 15);
}
// B-fragment: same K grouping as A ({0-7,16-23}/{8-15,24-31})
static __device__ __forceinline__ v16h load_b_frag(const _Float16* p) {
  v8h lo = *(const v8h*)(p);
  v8h hi = *(const v8h*)(p + 16);
  return __builtin_shufflevector(lo, hi, 0, 1, 2, 3, 4, 5, 6, 7,
                                 8, 9, 10, 11, 12, 13, 14, 15);
}

// ---------------------------------------------------------------------------
// f32 -> f16 elementwise conversion (4 elements per thread)
// ---------------------------------------------------------------------------
__global__ __launch_bounds__(256) void cvt_f16_kernel(
    const float* __restrict__ src, _Float16* __restrict__ dst) {
  const size_t i4 = ((size_t)blockIdx.x * 256 + threadIdx.x) * 4;
  v4f x = *(const v4f*)(src + i4);
  v4h o;
  o[0] = (_Float16)x[0]; o[1] = (_Float16)x[1];
  o[2] = (_Float16)x[2]; o[3] = (_Float16)x[3];
  *(v4h*)(dst + i4) = o;
}

// ---------------------------------------------------------------------------
// Tensor Data Mover: 1-D tile of `nelem` f16 elements -> LDS (ISA D# layout)
// 6-arg builtin variant: (g0, g1, g2, g3, g4, cpol)
// ---------------------------------------------------------------------------
static __device__ __forceinline__ void tdm_load_f16_1d(
    unsigned int lds_byte_off, const void* gsrc, unsigned int nelem) {
  const unsigned long long ga = (unsigned long long)(uintptr_t)gsrc;
  ui32x4 g0;
  g0[0] = 1u;                                     // count=1 (valid user D#)
  g0[1] = lds_byte_off;                           // lds_addr
  g0[2] = (unsigned int)(ga & 0xffffffffu);       // global_addr[31:0]
  g0[3] = (unsigned int)((ga >> 32) & 0x01ffffffu) | (2u << 30);  // type=2
  i32x8 g1;
  g1[0] = (int)(1u << 16);                        // data_size=1 (2 bytes)
  g1[1] = (int)((nelem & 0xffffu) << 16);         // tensor_dim0[15:0]
  g1[2] = (int)(((nelem >> 16) & 0xffffu) | (1u << 16)); // dim0 hi | tensor_dim1=1
  g1[3] = (int)((nelem & 0xffffu) << 16);         // tile_dim0
  g1[4] = 0;                                      // tile_dim1/2 unused
  g1[5] = (int)nelem;                             // tensor_dim0_stride lo
  g1[6] = 0;
  g1[7] = 0;
  i32x4 g2 = {0, 0, 0, 0};
  i32x4 g3 = {0, 0, 0, 0};
  i32x8 g4 = {0, 0, 0, 0, 0, 0, 0, 0};
  __builtin_amdgcn_tensor_load_to_lds(g0, g1, g2, g3, g4, 0);
}

// ---------------------------------------------------------------------------
// QKV projection: Y = X16 @ W16^T + bias (all-f16 operands, f32 accum)
// mode 0: store f16 as [B,H,S,DK]; mode 1: store f16 as [B,H,DK,S] (V^T)
// grid (NROW/128, D/64), block 256 (8 waves); wave: 16x64 output tile.
// ---------------------------------------------------------------------------
__global__ __launch_bounds__(256) void proj_qkv_kernel(
    const _Float16* __restrict__ X, const _Float16* __restrict__ W,
    const float* __restrict__ bias, _Float16* __restrict__ Y, int mode) {
  const int lane = threadIdx.x & 31;
  const int wave = threadIdx.x >> 5;
  const int half = lane >> 4;
  const int l16  = lane & 15;
  const int mtile = blockIdx.x * 8 + wave;
  const int nbase = blockIdx.y * 64;
  const int m = mtile * 16 + l16;

  v8f zero = {};
  v8f acc[4];
#pragma unroll
  for (int t = 0; t < 4; ++t) acc[t] = zero;

  const _Float16* xrow = X + (size_t)m * D_;
  const _Float16* wrow = W + (size_t)(nbase + l16) * D_;
#pragma unroll 2
  for (int kk = 0; kk < D_; kk += 32) {
    v16h a = load_a_frag(xrow + kk + half * 8);
    const int kb = kk + half * 8;
    v16h b0 = load_b_frag(wrow + kb);
    v16h b1 = load_b_frag(wrow + 16 * D_ + kb);
    v16h b2 = load_b_frag(wrow + 32 * D_ + kb);
    v16h b3 = load_b_frag(wrow + 48 * D_ + kb);
    acc[0] = wmma_f16(a, b0, acc[0]);
    acc[1] = wmma_f16(a, b1, acc[1]);
    acc[2] = wmma_f16(a, b2, acc[2]);
    acc[3] = wmma_f16(a, b3, acc[3]);
  }

#pragma unroll
  for (int t = 0; t < 4; ++t) {
    const int col = nbase + t * 16 + l16;
    const float bb = bias[col];
    const int hh = col / DK_;
    const int dk = col % DK_;
#pragma unroll
    for (int r = 0; r < 8; ++r) {
      const int row = mtile * 16 + r + 8 * half;
      const int bidx = row / S_;
      const int srow = row % S_;
      const float v = acc[t][r] + bb;
      size_t dst;
      if (mode == 0)
        dst = (((size_t)bidx * H_ + hh) * S_ + srow) * DK_ + dk;
      else
        dst = (((size_t)bidx * H_ + hh) * DK_ + dk) * S_ + srow;
      Y[dst] = (_Float16)v;
    }
  }
}

// ---------------------------------------------------------------------------
// logits + softmax; Q tile staged in LDS via Tensor Data Mover.
// grid (S/16, B*H), block 256 (8 waves); wave w covers keys [w*256, +256)
// ---------------------------------------------------------------------------
__global__ __launch_bounds__(256) void attn_softmax_kernel(
    const _Float16* __restrict__ Qh, const _Float16* __restrict__ Kh,
    float* __restrict__ attn) {
  const int lane = threadIdx.x & 31;
  const int wave = threadIdx.x >> 5;
  const int half = lane >> 4;
  const int l16  = lane & 15;
  const int bh    = blockIdx.y;
  const int qtile = blockIdx.x;
  const int kw    = wave * 256;

  __shared__ _Float16 sQ[16 * DK_];     // 2 KB Q tile
  __shared__ float    redbuf[8][16];

  if (threadIdx.x < 32) {
    const _Float16* qsrc = Qh + ((size_t)bh * S_ + qtile * 16) * DK_;
    tdm_load_f16_1d((unsigned int)(uintptr_t)(void*)sQ, qsrc, 16 * DK_);
    __builtin_amdgcn_s_wait_tensorcnt(0);
  }
  __syncthreads();

  // Q A-fragments from LDS (ds_load_b128 pairs)
  v16h qa[2];
#pragma unroll
  for (int f = 0; f < 2; ++f)
    qa[f] = load_a_frag(sQ + l16 * DK_ + f * 32 + half * 8);

  v8f zero = {};
  v8f acc[16];
#pragma unroll
  for (int t = 0; t < 16; ++t) acc[t] = zero;

#pragma unroll
  for (int t = 0; t < 16; ++t) {
    const int key = kw + t * 16 + l16;
    const _Float16* kp = Kh + ((size_t)bh * S_ + key) * DK_ + half * 8;
    v16h b0 = load_b_frag(kp);
    v16h b1 = load_b_frag(kp + 32);
    acc[t] = wmma_f16(qa[0], b0, acc[t]);
    acc[t] = wmma_f16(qa[1], b1, acc[t]);
  }

  float rstat[8];
  // row max: wave-local (shfl) then cross-wave (LDS)
#pragma unroll
  for (int r = 0; r < 8; ++r) {
    float mx = -1e30f;
#pragma unroll
    for (int t = 0; t < 16; ++t) mx = fmaxf(mx, acc[t][r]);
#pragma unroll
    for (int msk = 1; msk < 16; msk <<= 1) mx = fmaxf(mx, __shfl_xor(mx, msk, 32));
    rstat[r] = mx * SCALE_;
  }
  if (l16 == 0) {
#pragma unroll
    for (int r = 0; r < 8; ++r) redbuf[wave][r + 8 * half] = rstat[r];
  }
  __syncthreads();
  float gmax[8];
#pragma unroll
  for (int r = 0; r < 8; ++r) {
    float mx = -1e30f;
#pragma unroll
    for (int wv = 0; wv < 8; ++wv) mx = fmaxf(mx, redbuf[wv][r + 8 * half]);
    gmax[r] = mx;
  }
  __syncthreads();

  // exp + row sum
#pragma unroll
  for (int r = 0; r < 8; ++r) {
    float s = 0.0f;
#pragma unroll
    for (int t = 0; t < 16; ++t) {
      float p = __expf(acc[t][r] * SCALE_ - gmax[r]);
      acc[t][r] = p;
      s += p;
    }
#pragma unroll
    for (int msk = 1; msk < 16; msk <<= 1) s += __shfl_xor(s, msk, 32);
    rstat[r] = s;
  }
  if (l16 == 0) {
#pragma unroll
    for (int r = 0; r < 8; ++r) redbuf[wave][r + 8 * half] = rstat[r];
  }
  __syncthreads();
  float rinv[8];
#pragma unroll
  for (int r = 0; r < 8; ++r) {
    float s = 0.0f;
#pragma unroll
    for (int wv = 0; wv < 8; ++wv) s += redbuf[wv][r + 8 * half];
    rinv[r] = 1.0f / s;
  }

  float* arow = attn + ((size_t)bh * S_ + qtile * 16) * S_;
#pragma unroll
  for (int t = 0; t < 16; ++t) {
    const int key = kw + t * 16 + l16;
#pragma unroll
    for (int r = 0; r < 8; ++r) {
      const int qr = r + 8 * half;
      arow[(size_t)qr * S_ + key] = acc[t][r] * rinv[r];
    }
  }
}

// ---------------------------------------------------------------------------
// ctx = P @ V : grid (S/16, B*H), block 256 (8 waves split K=2048)
// ---------------------------------------------------------------------------
__global__ __launch_bounds__(256) void attn_pv_kernel(
    const float* __restrict__ attn, const _Float16* __restrict__ VhT,
    _Float16* __restrict__ Ctx) {
  const int lane = threadIdx.x & 31;
  const int wave = threadIdx.x >> 5;
  const int half = lane >> 4;
  const int l16  = lane & 15;
  const int bh    = blockIdx.y;
  const int qtile = blockIdx.x;
  const int b = bh / H_;
  const int h = bh % H_;

  v8f zero = {};
  v8f acc[4];
#pragma unroll
  for (int t = 0; t < 4; ++t) acc[t] = zero;

  const float* prow = attn + ((size_t)bh * S_ + qtile * 16 + l16) * S_;
  const _Float16* vrow = VhT + ((size_t)bh * DK_ + l16) * S_;
  for (int step = 0; step < 8; ++step) {
    const int kk = wave * 256 + step * 32;
    // probabilities f32 -> f16 A fragment
    const float* ap = prow + kk + half * 8;
    __builtin_prefetch((const void*)(ap + 32), 0, 0);
    v8f r0 = *(const v8f*)(ap);
    v8f r1 = *(const v8f*)(ap + 16);
    v16h a;
#pragma unroll
    for (int e = 0; e < 8; ++e) {
      a[e]     = (_Float16)r0[e];
      a[e + 8] = (_Float16)r1[e];
    }
    const int kb = kk + half * 8;
    v16h b0 = load_b_frag(vrow + kb);
    v16h b1 = load_b_frag(vrow + (size_t)16 * S_ + kb);
    v16h b2 = load_b_frag(vrow + (size_t)32 * S_ + kb);
    v16h b3 = load_b_frag(vrow + (size_t)48 * S_ + kb);
    acc[0] = wmma_f16(a, b0, acc[0]);
    acc[1] = wmma_f16(a, b1, acc[1]);
    acc[2] = wmma_f16(a, b2, acc[2]);
    acc[3] = wmma_f16(a, b3, acc[3]);
  }

  __shared__ float cbuf[8][16][64];
#pragma unroll
  for (int t = 0; t < 4; ++t)
#pragma unroll
    for (int r = 0; r < 8; ++r)
      cbuf[wave][r + 8 * half][t * 16 + l16] = acc[t][r];
  __syncthreads();

  const int tid = threadIdx.x;
#pragma unroll
  for (int i = 0; i < 4; ++i) {
    const int idx = tid * 4 + i;
    const int row = idx >> 6;
    const int col = idx & 63;
    float s = 0.0f;
#pragma unroll
    for (int wv = 0; wv < 8; ++wv) s += cbuf[wv][row][col];
    const int srow = qtile * 16 + row;
    Ctx[(((size_t)b * S_ + srow) * H_ + h) * DK_ + col] = (_Float16)s;
  }
}

// ---------------------------------------------------------------------------
// out = ctx @ Wo^T + bo  (ctx f16, Wo16 f16, out f32)
// ---------------------------------------------------------------------------
__global__ __launch_bounds__(256) void out_proj_kernel(
    const _Float16* __restrict__ Xh, const _Float16* __restrict__ W,
    const float* __restrict__ bias, float* __restrict__ Y) {
  const int lane = threadIdx.x & 31;
  const int wave = threadIdx.x >> 5;
  const int half = lane >> 4;
  const int l16  = lane & 15;
  const int mtile = blockIdx.x * 8 + wave;
  const int nbase = blockIdx.y * 64;
  const int m = mtile * 16 + l16;

  v8f zero = {};
  v8f acc[4];
#pragma unroll
  for (int t = 0; t < 4; ++t) acc[t] = zero;

  const _Float16* xrow = Xh + (size_t)m * D_;
  const _Float16* wrow = W + (size_t)(nbase + l16) * D_;
#pragma unroll 2
  for (int kk = 0; kk < D_; kk += 32) {
    v16h a = load_a_frag(xrow + kk + half * 8);
    const int kb = kk + half * 8;
    v16h b0 = load_b_frag(wrow + kb);
    v16h b1 = load_b_frag(wrow + 16 * D_ + kb);
    v16h b2 = load_b_frag(wrow + 32 * D_ + kb);
    v16h b3 = load_b_frag(wrow + 48 * D_ + kb);
    acc[0] = wmma_f16(a, b0, acc[0]);
    acc[1] = wmma_f16(a, b1, acc[1]);
    acc[2] = wmma_f16(a, b2, acc[2]);
    acc[3] = wmma_f16(a, b3, acc[3]);
  }

#pragma unroll
  for (int t = 0; t < 4; ++t) {
    const int col = nbase + t * 16 + l16;
    const float bb = bias[col];
#pragma unroll
    for (int r = 0; r < 8; ++r) {
      const int row = mtile * 16 + r + 8 * half;
      Y[(size_t)row * D_ + col] = acc[t][r] + bb;
    }
  }
}

// ---------------------------------------------------------------------------
extern "C" void kernel_launch(void* const* d_in, const int* in_sizes, int n_in,
                              void* d_out, int out_size, void* d_ws,
                              size_t ws_size, hipStream_t stream) {
  (void)in_sizes; (void)n_in; (void)out_size; (void)ws_size;
  const float* q  = (const float*)d_in[0];
  const float* k  = (const float*)d_in[1];
  const float* v  = (const float*)d_in[2];
  const float* Wq = (const float*)d_in[3];
  const float* bq = (const float*)d_in[4];
  const float* Wk = (const float*)d_in[5];
  const float* bk = (const float*)d_in[6];
  const float* Wv = (const float*)d_in[7];
  const float* bv = (const float*)d_in[8];
  const float* Wo = (const float*)d_in[9];
  const float* bo = (const float*)d_in[10];

  float* outp = (float*)d_out;
  float* attn = outp + (size_t)B_ * S_ * D_;

  const size_t NX = (size_t)B_ * S_ * D_;   // 8,388,608
  const size_t NW = (size_t)D_ * D_;        // 1,048,576
  _Float16* Xq16 = (_Float16*)d_ws;
  _Float16* Xk16 = Xq16 + NX;
  _Float16* Xv16 = Xk16 + NX;
  _Float16* Wq16 = Xv16 + NX;
  _Float16* Wk16 = Wq16 + NW;
  _Float16* Wv16 = Wk16 + NW;
  _Float16* Wo16 = Wv16 + NW;
  _Float16* Qh   = Wo16 + NW;
  _Float16* Kh   = Qh + NX;
  _Float16* VhT  = Kh + NX;
  _Float16* Ctx  = VhT + NX;

  const int gX = (int)(NX / 4 / 256);  // 8192 blocks
  const int gW = (int)(NW / 4 / 256);  // 1024 blocks
  cvt_f16_kernel<<<gX, 256, 0, stream>>>(q, Xq16);
  cvt_f16_kernel<<<gX, 256, 0, stream>>>(k, Xk16);
  cvt_f16_kernel<<<gX, 256, 0, stream>>>(v, Xv16);
  cvt_f16_kernel<<<gW, 256, 0, stream>>>(Wq, Wq16);
  cvt_f16_kernel<<<gW, 256, 0, stream>>>(Wk, Wk16);
  cvt_f16_kernel<<<gW, 256, 0, stream>>>(Wv, Wv16);
  cvt_f16_kernel<<<gW, 256, 0, stream>>>(Wo, Wo16);

  dim3 gProj(NROW_ / 128, D_ / 64);   // (64, 16)
  proj_qkv_kernel<<<gProj, 256, 0, stream>>>(Xq16, Wq16, bq, Qh, 0);
  proj_qkv_kernel<<<gProj, 256, 0, stream>>>(Xk16, Wk16, bk, Kh, 0);
  proj_qkv_kernel<<<gProj, 256, 0, stream>>>(Xv16, Wv16, bv, VhT, 1);

  dim3 gAttn(S_ / 16, B_ * H_);       // (128, 64)
  attn_softmax_kernel<<<gAttn, 256, 0, stream>>>(Qh, Kh, attn);
  attn_pv_kernel<<<gAttn, 256, 0, stream>>>(attn, VhT, Ctx);

  out_proj_kernel<<<gProj, 256, 0, stream>>>(Ctx, Wo16, bo, outp);
}
